// EdgeGATv2Layer_45947560133204
// MI455X (gfx1250) — matrix operand order repaired and checked by reading, not compiled, not run on hardware
//
#include <hip/hip_runtime.h>
#include <hip/hip_bf16.h>

// ---- problem constants (match reference) ----
#define NN      100000
#define EE      1600000
#define IN_DIM  128
#define OUT_DIM 64
#define EDGE_DIM 32
#define NH      4
#define HD      16
#define NEG_SLOPE 0.2f

typedef __attribute__((ext_vector_type(16))) __bf16 v16bf;
typedef __attribute__((ext_vector_type(8)))  float  v8f;

// ---------------------------------------------------------------------------
// Kernel W: one-shot fp32 -> bf16 weight conversion (weights stay L2-resident;
// removes per-block cvt VALU work and halves weight fetch bytes in the GEMMs)
// ---------------------------------------------------------------------------
__global__ __launch_bounds__(256) void wconv_kernel(
    const float* __restrict__ W_src,
    const float* __restrict__ W_dst,
    const float* __restrict__ W_edge,
    __bf16* __restrict__ Wsb,
    __bf16* __restrict__ Wdb,
    __bf16* __restrict__ Web)
{
    int i = blockIdx.x * 256 + threadIdx.x;
    if (i < OUT_DIM * IN_DIM) {
        Wsb[i] = (__bf16)W_src[i];
        Wdb[i] = (__bf16)W_dst[i];
    }
    if (i < OUT_DIM * EDGE_DIM) Web[i] = (__bf16)W_edge[i];
}

// ---------------------------------------------------------------------------
// Kernel 0: init rst = bias, seg_max = -inf, seg_sum = 0
// ---------------------------------------------------------------------------
__global__ __launch_bounds__(256) void init_kernel(
    const float* __restrict__ bias,
    float* __restrict__ rst,
    float* __restrict__ seg_max,
    float* __restrict__ seg_sum)
{
    long i = (long)blockIdx.x * 256 + threadIdx.x;
    if (i < (long)NN * OUT_DIM) rst[i] = bias[i & 63];
    if (i < (long)NN * NH) { seg_max[i] = -__builtin_inff(); seg_sum[i] = 0.0f; }
}

// ---------------------------------------------------------------------------
// Kernel 1: node projections  feat_src = X @ W_src^T, feat_dst = X @ W_dst^T
// Block = 256 (8 waves). Each block: 16 node rows. Wave w: matrix (w/4),
// column tile (w%4). A (16x128 bf16) staged in LDS once per block.
// WMMA bf16 16x16x32, 4 K-steps; B loaded pre-converted bf16 (contiguous
// 16-K runs per lane -> merged b128 loads, no cvt in the loop).
// ---------------------------------------------------------------------------
__global__ __launch_bounds__(256) void node_proj_kernel(
    const float* __restrict__ node_feat,
    const __bf16* __restrict__ Wsb,
    const __bf16* __restrict__ Wdb,
    float* __restrict__ feat_src,
    float* __restrict__ feat_dst)
{
    __shared__ __bf16 lds_a[16 * IN_DIM];
    const int tid  = threadIdx.x;
    const long row0 = (long)blockIdx.x * 16;

    // stage + convert A slab: 16x128 fp32 -> bf16 (coalesced)
    #pragma unroll
    for (int i = 0; i < 8; ++i) {
        int idx = i * 256 + tid;                       // 0..2047
        lds_a[idx] = (__bf16)node_feat[row0 * IN_DIM + idx];
    }
    __syncthreads();

    const int wave = tid >> 5;
    const int lane = tid & 31;
    const int mat  = wave >> 2;        // 0 = W_src, 1 = W_dst
    const int c0   = (wave & 3) * 16;  // output column tile
    const int l16  = lane & 15;
    const int hi   = lane >> 4;
    const __bf16* __restrict__ W    = mat ? Wdb : Wsb;
    float*        __restrict__ outp = mat ? feat_dst : feat_src;

    v8f acc = {};
    #pragma unroll
    for (int k0 = 0; k0 < IN_DIM; k0 += 32) {
        v16bf a, b;
        // A k order used here (contiguous, same as B): lane m = l16;
        // VGPR j<4 : K = 16*hi + 2j ; VGPR j>=4 : K = 16*hi + 8 + 2(j-4)
        #pragma unroll
        for (int j = 0; j < 4; ++j) {
            int ka = k0 + 16 * hi + 2 * j;
            a[2 * j]     = lds_a[l16 * IN_DIM + ka];
            a[2 * j + 1] = lds_a[l16 * IN_DIM + ka + 1];
            int kb = k0 + 16 * hi + 8 + 2 * j;
            a[2 * (j + 4)]     = lds_a[l16 * IN_DIM + kb];
            a[2 * (j + 4) + 1] = lds_a[l16 * IN_DIM + kb + 1];
        }
        // B k order used here (contiguous, same as A): lane n = l16,
        // lanes 0-15 hold K=0..15, lanes 16-31 hold K=16..31 (2 per VGPR).
        // B[k][n] = W[n][k]  (B = W^T)
        const int n = c0 + l16;
        #pragma unroll
        for (int j = 0; j < 8; ++j) {
            int k = k0 + 16 * hi + 2 * j;
            b[2 * j]     = W[n * IN_DIM + k];
            b[2 * j + 1] = W[n * IN_DIM + k + 1];
        }
        acc = __builtin_amdgcn_wmma_f32_16x16x32_bf16(
            false, a, false, b, (short)0, acc, false, false);
    }
    // D layout: VGPR v -> M = v + 8*hi ; N = l16
    #pragma unroll
    for (int v = 0; v < 8; ++v) {
        outp[(row0 + v + 8 * hi) * OUT_DIM + c0 + l16] = acc[v];
    }
}

// ---------------------------------------------------------------------------
// Kernel 2: fused edge projection + gather + leaky-relu + attention dot.
// Block = 256 (8 waves), 32 edges/block. Wave w: edge group (w/4), head (w%4).
// One WMMA (K = EDGE_DIM = 32) per wave; 16-col tile == one head.
// Cross-lane reduce over HD (lane%16) via shfl_xor; write logits + atomicMax.
// ---------------------------------------------------------------------------
__global__ __launch_bounds__(256) void edge_logits_kernel(
    const float*  __restrict__ edge_feat,
    const __bf16* __restrict__ Web,
    const float*  __restrict__ attn,
    const float*  __restrict__ feat_src,
    const float*  __restrict__ feat_dst,
    const int*    __restrict__ src,
    const int*    __restrict__ dst,
    float* __restrict__ logits,
    float* __restrict__ seg_max)
{
    __shared__ __bf16 lds_e[32 * EDGE_DIM];
    const int tid = threadIdx.x;
    const long e0blk = (long)blockIdx.x * 32;

    #pragma unroll
    for (int i = 0; i < 4; ++i) {
        int idx = i * 256 + tid;                       // 0..1023
        lds_e[idx] = (__bf16)edge_feat[e0blk * EDGE_DIM + idx];
    }
    __syncthreads();

    const int wave = tid >> 5;
    const int lane = tid & 31;
    const int eg   = wave >> 2;        // edge sub-group (0/1)
    const int h    = wave & 3;         // head == column tile
    const int l16  = lane & 15;
    const int hi   = lane >> 4;
    const long e0  = e0blk + eg * 16;

    v16bf a, b;
    const int mrowbase = (eg * 16 + l16) * EDGE_DIM;
    #pragma unroll
    for (int j = 0; j < 4; ++j) {
        int ka = 16 * hi + 2 * j;
        a[2 * j]     = lds_e[mrowbase + ka];
        a[2 * j + 1] = lds_e[mrowbase + ka + 1];
        int kb = 16 * hi + 8 + 2 * j;
        a[2 * (j + 4)]     = lds_e[mrowbase + kb];
        a[2 * (j + 4) + 1] = lds_e[mrowbase + kb + 1];
    }
    const int n = h * 16 + l16;        // output column (== h*HD + hd)
    #pragma unroll
    for (int j = 0; j < 8; ++j) {
        int k = 16 * hi + 2 * j;
        b[2 * j]     = Web[n * EDGE_DIM + k];
        b[2 * j + 1] = Web[n * EDGE_DIM + k + 1];
    }
    v8f acc = {};
    acc = __builtin_amdgcn_wmma_f32_16x16x32_bf16(
        false, a, false, b, (short)0, acc, false, false);

    const float aw = attn[h * HD + l16];
    #pragma unroll
    for (int v = 0; v < 8; ++v) {
        long e = e0 + v + 8 * hi;      // D layout: M = v + 8*hi
        int s = src[e], d = dst[e];
        float val = acc[v]
                  + feat_src[(long)s * OUT_DIM + n]
                  + feat_dst[(long)d * OUT_DIM + n];
        val = val > 0.0f ? val : NEG_SLOPE * val;      // GATv2 leaky-relu
        float t = val * aw;
        // reduce over HD: lanes within each 16-lane group
        t += __shfl_xor(t, 1, 16);
        t += __shfl_xor(t, 2, 16);
        t += __shfl_xor(t, 4, 16);
        t += __shfl_xor(t, 8, 16);
        if (l16 == 0) {
            logits[e * NH + h] = t;
            __hip_atomic_fetch_max(&seg_max[(long)d * NH + h], t,
                                   __ATOMIC_RELAXED, __HIP_MEMORY_SCOPE_AGENT);
        }
    }
}

// ---------------------------------------------------------------------------
// Kernel 3: ex = exp(logit - seg_max[dst]) (in place), seg_sum += ex
// ---------------------------------------------------------------------------
__global__ __launch_bounds__(256) void softmax_ex_kernel(
    const int*   __restrict__ dst,
    const float* __restrict__ seg_max,
    float* __restrict__ logits_ex,
    float* __restrict__ seg_sum)
{
    long i = (long)blockIdx.x * 256 + threadIdx.x;
    if (i >= (long)EE * NH) return;
    long e = i >> 2; int h = (int)(i & 3);
    int d = dst[e];
    float ex = __expf(logits_ex[i] - seg_max[(long)d * NH + h]);
    logits_ex[i] = ex;
    atomicAdd(&seg_sum[(long)d * NH + h], ex);
}

// ---------------------------------------------------------------------------
// Kernel 4: alpha = ex/seg_sum[dst] -> output; rst[dst] += feat_src[src]*alpha
// Thread per (edge, out-col): coalesced row reads, L2-resident atomics.
// ---------------------------------------------------------------------------
__global__ __launch_bounds__(256) void aggregate_kernel(
    const int*   __restrict__ src,
    const int*   __restrict__ dst,
    const float* __restrict__ feat_src,
    const float* __restrict__ ex,
    const float* __restrict__ seg_sum,
    float* __restrict__ rst,
    float* __restrict__ alpha_out)
{
    long i = (long)blockIdx.x * 256 + threadIdx.x;
    if (i >= (long)EE * OUT_DIM) return;
    long e = i >> 6; int c = (int)(i & 63); int h = c >> 4;
    int d = dst[e], s = src[e];
    float al = ex[e * NH + h] / seg_sum[(long)d * NH + h];
    if ((c & 15) == 0) alpha_out[e * NH + h] = al;
    atomicAdd(&rst[(long)d * OUT_DIM + c],
              feat_src[(long)s * OUT_DIM + c] * al);
}

// ---------------------------------------------------------------------------
extern "C" void kernel_launch(void* const* d_in, const int* in_sizes, int n_in,
                              void* d_out, int out_size, void* d_ws, size_t ws_size,
                              hipStream_t stream) {
    const float* node_feat = (const float*)d_in[0];
    const float* edge_feat = (const float*)d_in[1];
    const float* W_src     = (const float*)d_in[2];
    const float* W_dst     = (const float*)d_in[3];
    const float* W_edge    = (const float*)d_in[4];
    const float* attn      = (const float*)d_in[5];
    const float* bias      = (const float*)d_in[6];
    const int*   src       = (const int*)d_in[7];
    const int*   dst       = (const int*)d_in[8];

    float* rst       = (float*)d_out;                      // [N, 64]
    float* alpha_out = (float*)d_out + (long)NN * OUT_DIM; // [E, 4]

    // workspace layout (floats then bf16): ~80 MB total
    float* ws       = (float*)d_ws;
    float* feat_src = ws;                                  // N*64
    float* feat_dst = feat_src + (long)NN * OUT_DIM;       // N*64
    float* logits   = feat_dst + (long)NN * OUT_DIM;       // E*4 (becomes ex)
    float* seg_max  = logits   + (long)EE * NH;            // N*4
    float* seg_sum  = seg_max  + (long)NN * NH;            // N*4
    __bf16* Wsb = (__bf16*)(seg_sum + (long)NN * NH);      // 64*128 bf16
    __bf16* Wdb = Wsb + OUT_DIM * IN_DIM;                  // 64*128 bf16
    __bf16* Web = Wdb + OUT_DIM * IN_DIM;                  // 64*32  bf16

    const int B = 256;
    wconv_kernel<<<(OUT_DIM * IN_DIM + B - 1) / B, B, 0, stream>>>(
        W_src, W_dst, W_edge, Wsb, Wdb, Web);
    init_kernel<<<(NN * OUT_DIM + B - 1) / B, B, 0, stream>>>(
        bias, rst, seg_max, seg_sum);
    node_proj_kernel<<<NN / 16, B, 0, stream>>>(
        node_feat, Wsb, Wdb, feat_src, feat_dst);
    edge_logits_kernel<<<EE / 32, B, 0, stream>>>(
        edge_feat, Web, attn, feat_src, feat_dst, src, dst, logits, seg_max);
    softmax_ex_kernel<<<(int)(((long)EE * NH + B - 1) / B), B, 0, stream>>>(
        dst, seg_max, logits, seg_sum);
    aggregate_kernel<<<(int)(((long)EE * OUT_DIM + B - 1) / B), B, 0, stream>>>(
        src, dst, feat_src, logits, seg_sum, rst, alpha_out);
}
